// GPT2_6167573037528
// MI455X (gfx1250) — compile-verified, matrix operand order checked by reading, not run on hardware
//
#include <hip/hip_runtime.h>
#include <hip/hip_bf16.h>

// ---------------------------------------------------------------------------
// GPT-2 (124M) forward on gfx1250: WMMA f16 GEMMs with f32 accumulation,
// async global->LDS tile staging (CDNA5 ASYNCcnt path).
// ---------------------------------------------------------------------------

#define V_VOCAB 50257
#define NL 12
#define NH 12
#define CH 768
#define BB 2
#define TT 1024
#define DH 64
#define MROWS (BB * TT)          // 2048
#define LN_EPS 1e-5f
#define GELU_K 0.7978845608028654f

typedef _Float16 half_t;
typedef __attribute__((ext_vector_type(16))) _Float16 v16h;
typedef __attribute__((ext_vector_type(8)))  float    v8f;

struct __align__(16) U4 { unsigned int x, y, z, w; };
struct __align__(16) F4 { float x, y, z, w; };

union FragH { v16h v; U4 q[2]; };

__device__ __forceinline__ v8f wmma_f16(v16h a, v16h b, v8f c) {
  return __builtin_amdgcn_wmma_f32_16x16x32_f16(
      /*neg_a=*/false, a, /*neg_b=*/false, b,
      /*c_mod=*/(short)0, c, /*reuse_a=*/false, /*reuse_b=*/false);
}

// Async copy of 16 bytes global -> LDS (per lane). Wave-relative LDS offset
// is the low 32 bits of the flat shared-memory address (flat LDS aperture
// maps addr[31:0] directly to the wave's LDS allocation).
__device__ __forceinline__ void async_copy_b128(void* lds_dst, const void* gsrc) {
  unsigned lds_off = (unsigned)(unsigned long long)lds_dst;
  asm volatile("global_load_async_to_lds_b128 %0, %1, off"
               :: "v"(lds_off), "v"(gsrc) : "memory");
}

__device__ __forceinline__ void async_wait_all() {
  asm volatile("s_wait_asynccnt 0x0" ::: "memory");
}

__device__ __forceinline__ float gelu_f(float x) {
  float x3 = x * x * x;
  return 0.5f * x * (1.0f + tanhf(GELU_K * (x + 0.044715f * x3)));
}

__device__ __forceinline__ float block_reduce_sum(float v, float* red, int slot) {
#pragma unroll
  for (int m = 1; m < 32; m <<= 1) v += __shfl_xor(v, m, 32);
  int wave = threadIdx.x >> 5;
  int lane = threadIdx.x & 31;
  if (lane == 0) red[wave] = v;
  __syncthreads();
  if (threadIdx.x == 0) {
    float t = 0.0f;
    int nw = blockDim.x >> 5;
    for (int i = 0; i < nw; ++i) t += red[i];
    red[slot] = t;
  }
  __syncthreads();
  return red[slot];
}

__device__ __forceinline__ float block_reduce_max(float v, float* red, int slot) {
#pragma unroll
  for (int m = 1; m < 32; m <<= 1) v = fmaxf(v, __shfl_xor(v, m, 32));
  int wave = threadIdx.x >> 5;
  int lane = threadIdx.x & 31;
  if (lane == 0) red[wave] = v;
  __syncthreads();
  if (threadIdx.x == 0) {
    float t = -3.0e38f;
    int nw = blockDim.x >> 5;
    for (int i = 0; i < nw; ++i) t = fmaxf(t, red[i]);
    red[slot] = t;
  }
  __syncthreads();
  return red[slot];
}

// ---------------------------------------------------------------------------
// Embedding: x[b*T+t, c] = wte[idx[b,t], c] + wpe[t, c]
// ---------------------------------------------------------------------------
__global__ __launch_bounds__(256) void gpt2_embed(
    const int* __restrict__ idx, const float* __restrict__ wte,
    const float* __restrict__ wpe, float* __restrict__ x)
{
  int row = blockIdx.x;            // 0..MROWS-1
  int t = row % TT;
  int tok = idx[row];
  const float* we = wte + (size_t)tok * CH;
  const float* pe = wpe + (size_t)t * CH;
  float* xr = x + (size_t)row * CH;
  for (int c = threadIdx.x; c < CH; c += 256) xr[c] = we[c] + pe[c];
}

// ---------------------------------------------------------------------------
// LayerNorm: out_f16[row, c] = (x - mean)/sqrt(var + eps) * w + b
// ---------------------------------------------------------------------------
__global__ __launch_bounds__(256) void gpt2_layernorm(
    const float* __restrict__ x, const float* __restrict__ w,
    const float* __restrict__ b, half_t* __restrict__ out)
{
  __shared__ float red[12];
  int row = blockIdx.x;
  const float* xr = x + (size_t)row * CH;
  float s = 0.0f;
  for (int c = threadIdx.x; c < CH; c += 256) s += xr[c];
  float mean = block_reduce_sum(s, red, 8) * (1.0f / CH);
  float ss = 0.0f;
  for (int c = threadIdx.x; c < CH; c += 256) {
    float d = xr[c] - mean;
    ss += d * d;
  }
  float var = block_reduce_sum(ss, red, 9) * (1.0f / CH);
  float rstd = rsqrtf(var + LN_EPS);
  half_t* orow = out + (size_t)row * CH;
  for (int c = threadIdx.x; c < CH; c += 256)
    orow[c] = (half_t)((xr[c] - mean) * rstd * w[c] + b[c]);
}

// ---------------------------------------------------------------------------
// GEMM: out[M,N] = A_f16[M,K] @ W_f32 (+ bias)(gelu)(+ res), f32 or f16 out.
// transB==0: W is [K,N] row-major. transB==1: W is [N,K] row-major (W^T).
// Block tile 128x128x32; 8 waves, each wave 32x64 (2x4 WMMA 16x16 tiles).
// A tile staged via GLOBAL_LOAD_ASYNC_TO_LDS_B128 (ASYNCcnt); B tile is
// converted f32->f16 on the VALU path.
// ---------------------------------------------------------------------------
#define GBM 128
#define GBN 128
#define GBK 32
#define ASTR 40
#define BSTR 40

__global__ __launch_bounds__(256) void gpt2_gemm(
    const half_t* __restrict__ A, const float* __restrict__ W,
    const float* __restrict__ bias, const float* __restrict__ res,
    float* __restrict__ outF, half_t* __restrict__ outH,
    int M, int N, int K, int transB, int act)
{
  __shared__ __align__(16) half_t Alds[GBM][ASTR];
  __shared__ __align__(16) half_t Blds[GBN][BSTR];   // transposed: [n][k]

  int tid = threadIdx.x;
  int lane = tid & 31;
  int wave = tid >> 5;
  int wm = wave & 3;               // 0..3, M direction
  int wn = wave >> 2;              // 0..1, N direction
  int row16 = lane & 15;
  int hsel = lane >> 4;            // 0/1: which K/M half per operand layout
  int m0 = blockIdx.y * GBM;
  int n0 = blockIdx.x * GBN;

  v8f acc[2][4] = {};

  for (int k0 = 0; k0 < K; k0 += GBK) {
    __syncthreads();
    // --- stage A tile (f16) via async DMA : 128 x 32 ---
#pragma unroll
    for (int it = 0; it < 2; ++it) {
      int i = tid + it * 256;          // 0..511
      int r = i >> 2;                  // 0..127
      int g = i & 3;                   // 8-half group
      async_copy_b128(&Alds[r][g * 8],
                      &A[(size_t)(m0 + r) * K + k0 + g * 8]);
    }
    // --- stage B tile (convert f32 -> f16, store transposed [n][k]) ---
    if (!transB) {
#pragma unroll
      for (int it = 0; it < 4; ++it) {
        int i = tid + it * 256;        // 0..1023
        int k = i >> 5;                // 0..31
        int n = (i & 31) * 4;          // 0..124
        F4 wv = *(const F4*)&W[(size_t)(k0 + k) * N + n0 + n];
        Blds[n + 0][k] = (half_t)wv.x;
        Blds[n + 1][k] = (half_t)wv.y;
        Blds[n + 2][k] = (half_t)wv.z;
        Blds[n + 3][k] = (half_t)wv.w;
      }
    } else {
#pragma unroll
      for (int it = 0; it < 4; ++it) {
        int i = tid + it * 256;        // 0..1023
        int n = i >> 3;                // 0..127
        int k = (i & 7) * 4;           // 0..28
        F4 wv; wv.x = wv.y = wv.z = wv.w = 0.0f;
        if (n0 + n < N)
          wv = *(const F4*)&W[(size_t)(n0 + n) * K + k0 + k];
        Blds[n][k + 0] = (half_t)wv.x;
        Blds[n][k + 1] = (half_t)wv.y;
        Blds[n][k + 2] = (half_t)wv.z;
        Blds[n][k + 3] = (half_t)wv.w;
      }
    }
    async_wait_all();
    __syncthreads();

    // --- fragments + WMMA ---
    FragH af[2], bf[4];
#pragma unroll
    for (int mi = 0; mi < 2; ++mi) {
      int r = wm * 32 + mi * 16 + row16;
      af[mi].q[0] = *(const U4*)&Alds[r][hsel * 8];
      af[mi].q[1] = *(const U4*)&Alds[r][16 + hsel * 8];
    }
#pragma unroll
    for (int ni = 0; ni < 4; ++ni) {
      int c = wn * 64 + ni * 16 + row16;
      bf[ni].q[0] = *(const U4*)&Blds[c][hsel * 8];
      bf[ni].q[1] = *(const U4*)&Blds[c][16 + hsel * 8];
    }
#pragma unroll
    for (int mi = 0; mi < 2; ++mi)
#pragma unroll
      for (int ni = 0; ni < 4; ++ni)
        acc[mi][ni] = wmma_f16(af[mi].v, bf[ni].v, acc[mi][ni]);
  }

  // --- epilogue: bias / gelu / residual / store ---
#pragma unroll
  for (int mi = 0; mi < 2; ++mi) {
#pragma unroll
    for (int ni = 0; ni < 4; ++ni) {
#pragma unroll
      for (int r = 0; r < 8; ++r) {
        int m = m0 + wm * 32 + mi * 16 + r + hsel * 8;
        int n = n0 + wn * 64 + ni * 16 + row16;
        if (n >= N) continue;
        float v = acc[mi][ni][r];
        if (bias) v += bias[n];
        if (act == 1) v = gelu_f(v);
        size_t o = (size_t)m * N + n;
        if (res) v += res[o];
        if (outF) outF[o] = v;
        else      outH[o] = (half_t)v;
      }
    }
  }
}

// ---------------------------------------------------------------------------
// Causal attention (flash-style, online softmax) over qkv_f16 [M, 3C].
// Grid: (B*H, T/64). Block: 128 threads = 4 waves, 16 query rows per wave.
// Key blocks of 32; S = Q K^T (2 WMMAs / 16-key tile), P V (4 WMMAs).
// K tile staged via async DMA; V staged transposed on the VALU path.
// ---------------------------------------------------------------------------
#define QB 64
#define KBLK 32
#define KSTR 72
#define VSTR 40
#define PSTR 40

__global__ __launch_bounds__(128) void gpt2_attention(
    const half_t* __restrict__ qkv, half_t* __restrict__ y)
{
  __shared__ __align__(16) half_t Klds[KBLK][KSTR];       // [key][d]
  __shared__ __align__(16) half_t VldsT[DH][VSTR];        // [d][key]
  __shared__ __align__(16) half_t Plds[4][16][PSTR];      // per-wave P tile

  int tid = threadIdx.x;
  int lane = tid & 31;
  int wave = tid >> 5;
  int row16 = lane & 15;
  int hsel = lane >> 4;
  int b = blockIdx.x / NH;
  int h = blockIdx.x % NH;
  int qblk = blockIdx.y;
  int q0 = qblk * QB + wave * 16;

  // Q fragments (A operand, d = 0..31 and 32..63)
  FragH qa0, qa1;
  {
    const half_t* qp = qkv + (size_t)(b * TT + q0 + row16) * (3 * CH) + h * DH;
    qa0.q[0] = *(const U4*)&qp[hsel * 8];
    qa0.q[1] = *(const U4*)&qp[16 + hsel * 8];
    qa1.q[0] = *(const U4*)&qp[32 + hsel * 8];
    qa1.q[1] = *(const U4*)&qp[48 + hsel * 8];
  }

  float mrun[8], lrun[8];
#pragma unroll
  for (int r = 0; r < 8; ++r) { mrun[r] = -1.0e30f; lrun[r] = 0.0f; }
  v8f o[4] = {};

  int nkb = qblk * 2 + 2;   // key blocks covering keys [0, qblk*64+64)
  for (int kb = 0; kb < nkb; ++kb) {
    __syncthreads();
    // --- stage K [32 x 64] (async) and V^T [64 x 32] (manual transpose) ---
#pragma unroll
    for (int it = 0; it < 2; ++it) {
      int i = tid + it * 128;        // 0..255
      int key = i >> 3;              // 0..31
      int g = i & 7;                 // d group (8 halfs)
      size_t base = (size_t)(b * TT + kb * KBLK + key) * (3 * CH) + h * DH;
      async_copy_b128(&Klds[key][g * 8], &qkv[base + CH + g * 8]);
      U4 vv = *(const U4*)&qkv[base + 2 * CH + g * 8];
      const half_t* vh = (const half_t*)&vv;
#pragma unroll
      for (int e = 0; e < 8; ++e) VldsT[g * 8 + e][key] = vh[e];
    }
    async_wait_all();
    __syncthreads();

    // --- S = Q @ K^T (scaled), two 16-key tiles ---
    v8f s[2];
#pragma unroll
    for (int nt = 0; nt < 2; ++nt) {
      FragH kf0, kf1;
      int keyc = nt * 16 + row16;
      kf0.q[0] = *(const U4*)&Klds[keyc][hsel * 8];
      kf0.q[1] = *(const U4*)&Klds[keyc][16 + hsel * 8];
      kf1.q[0] = *(const U4*)&Klds[keyc][32 + hsel * 8];
      kf1.q[1] = *(const U4*)&Klds[keyc][48 + hsel * 8];
      v8f z = {};
      z = wmma_f16(qa0.v, kf0.v, z);
      z = wmma_f16(qa1.v, kf1.v, z);
      s[nt] = z;
    }

    // --- causal mask + online softmax (rows split across 16-lane groups) ---
    float pv[2][8];
#pragma unroll
    for (int r = 0; r < 8; ++r) {
      int qpos = q0 + r + hsel * 8;
      float rmax = -1.0e30f;
#pragma unroll
      for (int nt = 0; nt < 2; ++nt) {
        int kpos = kb * KBLK + nt * 16 + row16;
        float sv = s[nt][r] * 0.125f;            // 1/sqrt(64)
        if (kpos > qpos) sv = -1.0e30f;
        pv[nt][r] = sv;
        rmax = fmaxf(rmax, sv);
      }
#pragma unroll
      for (int m = 1; m < 16; m <<= 1)
        rmax = fmaxf(rmax, __shfl_xor(rmax, m, 32));
      float mnew = fmaxf(mrun[r], rmax);
      float corr = expf(mrun[r] - mnew);
      float rsum = 0.0f;
#pragma unroll
      for (int nt = 0; nt < 2; ++nt) {
        float p = expf(pv[nt][r] - mnew);
        pv[nt][r] = p;
        rsum += p;
      }
#pragma unroll
      for (int m = 1; m < 16; m <<= 1)
        rsum += __shfl_xor(rsum, m, 32);
      lrun[r] = lrun[r] * corr + rsum;
      mrun[r] = mnew;
#pragma unroll
      for (int dt = 0; dt < 4; ++dt) o[dt][r] *= corr;
    }

    // --- restage P (C layout -> A layout) via LDS ---
    __syncthreads();
#pragma unroll
    for (int r = 0; r < 8; ++r) {
      int m = r + hsel * 8;
      Plds[wave][m][row16]      = (half_t)pv[0][r];
      Plds[wave][m][16 + row16] = (half_t)pv[1][r];
    }
    __syncthreads();
    FragH pa;
    pa.q[0] = *(const U4*)&Plds[wave][row16][hsel * 8];
    pa.q[1] = *(const U4*)&Plds[wave][row16][16 + hsel * 8];

    // --- O += P @ V ---
#pragma unroll
    for (int dt = 0; dt < 4; ++dt) {
      FragH vf;
      int d = dt * 16 + row16;
      vf.q[0] = *(const U4*)&VldsT[d][hsel * 8];
      vf.q[1] = *(const U4*)&VldsT[d][16 + hsel * 8];
      o[dt] = wmma_f16(pa.v, vf.v, o[dt]);
    }
  }

  // --- normalize + store y (f16, [M, C] interleaved by head) ---
#pragma unroll
  for (int r = 0; r < 8; ++r) {
    float inv = 1.0f / lrun[r];
    int m = q0 + r + hsel * 8;
#pragma unroll
    for (int dt = 0; dt < 4; ++dt)
      y[(size_t)(b * TT + m) * CH + h * DH + dt * 16 + row16] =
          (half_t)(o[dt][r] * inv);
  }
}

// ---------------------------------------------------------------------------
// Loss: per-row logsumexp + NLL, mean-accumulated via atomics.
// ---------------------------------------------------------------------------
__global__ void gpt2_zero_loss(float* __restrict__ loss) {
  if (blockIdx.x == 0 && threadIdx.x == 0) *loss = 0.0f;
}

__global__ __launch_bounds__(256) void gpt2_loss(
    const float* __restrict__ logits, const int* __restrict__ targets,
    float* __restrict__ loss)
{
  __shared__ float red[12];
  int row = blockIdx.x;
  const float* lr = logits + (size_t)row * V_VOCAB;
  float mx = -3.0e38f;
  for (int c = threadIdx.x; c < V_VOCAB; c += 256) mx = fmaxf(mx, lr[c]);
  mx = block_reduce_max(mx, red, 8);
  float se = 0.0f;
  for (int c = threadIdx.x; c < V_VOCAB; c += 256) se += expf(lr[c] - mx);
  se = block_reduce_sum(se, red, 9);
  if (threadIdx.x == 0) {
    float lt = lr[targets[row]];
    atomicAdd(loss, (mx + logf(se) - lt) * (1.0f / (float)MROWS));
  }
}

// ---------------------------------------------------------------------------
// Orchestration
// ---------------------------------------------------------------------------
static inline void launch_gemm(const half_t* A, const float* W,
                               const float* bias, const float* res,
                               float* oF, half_t* oH,
                               int M, int N, int K, int transB, int act,
                               hipStream_t s)
{
  dim3 grid((N + GBN - 1) / GBN, M / GBM);
  gpt2_gemm<<<grid, 256, 0, s>>>(A, W, bias, res, oF, oH, M, N, K, transB, act);
}

extern "C" void kernel_launch(void* const* d_in, const int* in_sizes, int n_in,
                              void* d_out, int out_size, void* d_ws, size_t ws_size,
                              hipStream_t stream) {
  (void)in_sizes; (void)n_in; (void)out_size; (void)ws_size;

  const float* wte     = (const float*)d_in[0];
  const float* wpe     = (const float*)d_in[1];
  const float* ln1_w   = (const float*)d_in[2];
  const float* ln1_b   = (const float*)d_in[3];
  const float* attn_w  = (const float*)d_in[4];
  const float* attn_b  = (const float*)d_in[5];
  const float* attnp_w = (const float*)d_in[6];
  const float* attnp_b = (const float*)d_in[7];
  const float* ln2_w   = (const float*)d_in[8];
  const float* ln2_b   = (const float*)d_in[9];
  const float* fc_w    = (const float*)d_in[10];
  const float* fc_b    = (const float*)d_in[11];
  const float* mlpp_w  = (const float*)d_in[12];
  const float* mlpp_b  = (const float*)d_in[13];
  const float* lnf_w   = (const float*)d_in[14];
  const float* lnf_b   = (const float*)d_in[15];
  const int*   idx     = (const int*)d_in[16];
  const int*   targets = (const int*)d_in[17];

  float* out = (float*)d_out;                         // logits [M,V] then loss
  float* loss_ptr = out + (size_t)MROWS * V_VOCAB;

  // workspace layout
  char* ws = (char*)d_ws;
  size_t off = 0;
  float*  x    = (float*)(ws + off);  off += (size_t)MROWS * CH * 4;       // residual stream f32
  half_t* lnb  = (half_t*)(ws + off); off += (size_t)MROWS * CH * 2;       // LN output f16
  half_t* qkv  = (half_t*)(ws + off); off += (size_t)MROWS * 3 * CH * 2;   // QKV f16
  half_t* yb   = (half_t*)(ws + off); off += (size_t)MROWS * CH * 2;       // attn out f16
  half_t* hb   = (half_t*)(ws + off); off += (size_t)MROWS * 4 * CH * 2;   // GELU(fc) f16

  gpt2_embed<<<MROWS, 256, 0, stream>>>(idx, wte, wpe, x);

  for (int l = 0; l < NL; ++l) {
    const float* l_ln1w  = ln1_w   + (size_t)l * CH;
    const float* l_ln1b  = ln1_b   + (size_t)l * CH;
    const float* l_attnw = attn_w  + (size_t)l * CH * 3 * CH;
    const float* l_attnb = attn_b  + (size_t)l * 3 * CH;
    const float* l_apw   = attnp_w + (size_t)l * CH * CH;
    const float* l_apb   = attnp_b + (size_t)l * CH;
    const float* l_ln2w  = ln2_w   + (size_t)l * CH;
    const float* l_ln2b  = ln2_b   + (size_t)l * CH;
    const float* l_fcw   = fc_w    + (size_t)l * CH * 4 * CH;
    const float* l_fcb   = fc_b    + (size_t)l * 4 * CH;
    const float* l_mpw   = mlpp_w  + (size_t)l * 4 * CH * CH;
    const float* l_mpb   = mlpp_b  + (size_t)l * CH;

    gpt2_layernorm<<<MROWS, 256, 0, stream>>>(x, l_ln1w, l_ln1b, lnb);
    // qkv = ln1(x) @ attn_w + attn_b            (f16 out)
    launch_gemm(lnb, l_attnw, l_attnb, nullptr, nullptr, qkv,
                MROWS, 3 * CH, CH, 0, 0, stream);
    gpt2_attention<<<dim3(BB * NH, TT / QB), 128, 0, stream>>>(qkv, yb);
    // x = x + y @ attnp_w + attnp_b             (f32, fused residual)
    launch_gemm(yb, l_apw, l_apb, x, x, nullptr,
                MROWS, CH, CH, 0, 0, stream);
    gpt2_layernorm<<<MROWS, 256, 0, stream>>>(x, l_ln2w, l_ln2b, lnb);
    // h = gelu(ln2(x) @ fc_w + fc_b)            (f16 out, fused GELU)
    launch_gemm(lnb, l_fcw, l_fcb, nullptr, nullptr, hb,
                MROWS, 4 * CH, CH, 0, 1, stream);
    // x = x + h @ mlpp_w + mlpp_b               (f32, fused residual)
    launch_gemm(hb, l_mpw, l_mpb, x, x, nullptr,
                MROWS, CH, 4 * CH, 0, 0, stream);
  }

  gpt2_layernorm<<<MROWS, 256, 0, stream>>>(x, lnf_w, lnf_b, lnb);
  // logits = lnf(x) @ wte^T                     (transB, f32 out -> d_out)
  launch_gemm(lnb, wte, nullptr, nullptr, out, nullptr,
              MROWS, V_VOCAB, CH, 1, 0, stream);

  gpt2_zero_loss<<<1, 32, 0, stream>>>(loss_ptr);
  gpt2_loss<<<MROWS, 256, 0, stream>>>(out, targets, loss_ptr);
}
